// Head_32719060861057
// MI455X (gfx1250) — matrix operand order repaired and checked by reading, not compiled, not run on hardware
//
#include <hip/hip_runtime.h>
#include <hip/hip_bf16.h>

typedef __bf16 bf16_t;
typedef __attribute__((ext_vector_type(16))) __bf16 v16bf;
typedef __attribute__((ext_vector_type(8)))  __bf16 v8bf;
typedef __attribute__((ext_vector_type(8)))  float   v8f;
typedef __attribute__((ext_vector_type(4)))  unsigned int v4u;
typedef __attribute__((ext_vector_type(4)))  int v4i;
typedef __attribute__((ext_vector_type(8)))  int v8i;

typedef unsigned int       u32;
typedef unsigned long long u64;

union AF { v16bf v; v8bf h[2]; };
union F8 { v8f v; float f[8]; };

#define BB 8
#define TT 2048
#define DD 1024
#define HH 64

#if defined(__has_builtin)
#if __has_builtin(__builtin_amdgcn_tensor_load_to_lds) && \
    __has_builtin(__builtin_amdgcn_s_wait_tensorcnt)
#define HAVE_TDM 1
#endif
#endif

__device__ inline v8f wmma_bf16(v16bf a, v16bf b, v8f c) {
  // D = A(16x32 bf16) * B(32x16 bf16) + C(16x16 f32)
  return __builtin_amdgcn_wmma_f32_16x16x32_bf16(false, a, false, b, (short)0, c,
                                                 false, false);
}

#ifdef HAVE_TDM
// Issue a TDM 2-D tile load (bf16 elements) global -> LDS.
// tensor dims / strides / tile dims are in elements (data_size = 2 bytes).
__device__ inline void tdm_load_2d(u32 lds_off, const void* gptr,
                                   u32 tensor_d0, u32 tensor_d1,
                                   u32 tile_d0, u32 tile_d1, u32 stride0) {
  const u64 ga = (u64)(size_t)gptr;
  v4u g0;
  g0[0] = 1u;                                            // count=1, user D#
  g0[1] = lds_off;                                       // lds_addr (bytes)
  g0[2] = (u32)ga;                                       // global_addr[31:0]
  g0[3] = (u32)((ga >> 32) & 0x1FFFFFFu) | (2u << 30);   // addr[56:32] | type=2
  v8i g1;
  g1[0] = (int)(1u << 16);                               // data_size=1 (2B)
  g1[1] = (int)(tensor_d0 << 16);                        // tensor_dim0 lo
  g1[2] = (int)((tensor_d0 >> 16) | (tensor_d1 << 16));  // dim0 hi | dim1 lo
  g1[3] = (int)((tensor_d1 >> 16) | (tile_d0 << 16));    // dim1 hi | tile_dim0
  g1[4] = (int)(tile_d1 & 0xFFFFu);                      // tile_dim1 (dim2=0)
  g1[5] = (int)stride0;                                  // dim0_stride[31:0]
  g1[6] = 0;
  g1[7] = 0;
  v4i z4 = {0, 0, 0, 0};
#if __clang_major__ >= 23
  v8i z8 = {0, 0, 0, 0, 0, 0, 0, 0};
  __builtin_amdgcn_tensor_load_to_lds(g0, g1, z4, z4, z8, 0);
#else
  __builtin_amdgcn_tensor_load_to_lds(g0, g1, z4, z4, 0);
#endif
}
#endif  // HAVE_TDM

// ---------------------------------------------------------------------------
// prep: x (f32) -> xb (bf16); Wq/Wk/Wv (f32 [D,H]) -> wt (bf16 [3][H][D]).
// ---------------------------------------------------------------------------
__global__ void __launch_bounds__(256)
prep_kernel(const float* __restrict__ x, const float* __restrict__ Wq,
            const float* __restrict__ Wk, const float* __restrict__ Wv,
            bf16_t* __restrict__ xb, bf16_t* __restrict__ wt) {
  const long long nx = (long long)BB * TT * DD;
  const long long nw = 3LL * HH * DD;
  const long long total = nx + nw;
  for (long long i = (long long)blockIdx.x * blockDim.x + threadIdx.x; i < total;
       i += (long long)gridDim.x * blockDim.x) {
    if (i < nx) {
      xb[i] = (bf16_t)x[i];
    } else {
      long long j = i - nx;
      int p = (int)(j / (HH * DD));
      int rem = (int)(j % (HH * DD));
      int h = rem / DD, d = rem % DD;
      const float* W = (p == 0) ? Wq : ((p == 1) ? Wk : Wv);
      wt[j] = (bf16_t)W[(size_t)d * HH + h];
    }
  }
}

// ---------------------------------------------------------------------------
// proj: q,k,v = x @ W{q,k,v}. One wave per (16-token, 16-col) tile; the x
// A-fragment feeds 3 WMMAs. q,k row-major [B*T,H] bf16; v transposed [B,H,T].
// ---------------------------------------------------------------------------
__global__ void __launch_bounds__(128)
proj_kernel(const bf16_t* __restrict__ xb, const bf16_t* __restrict__ wt,
            bf16_t* __restrict__ qb, bf16_t* __restrict__ kb,
            bf16_t* __restrict__ vt) {
  const int wave = threadIdx.x >> 5;
  const int lane = threadIdx.x & 31;
  const int hf = lane >> 4;
  const int lm = lane & 15;
  const int row0 = blockIdx.x * 16;
  const int n0 = wave * 16;

  const bf16_t* aPtr = xb + (size_t)(row0 + lm) * DD + 8 * hf;
  const bf16_t* bq = wt + 0 * (size_t)HH * DD + (size_t)(n0 + lm) * DD + 8 * hf;
  const bf16_t* bk = wt + 1 * (size_t)HH * DD + (size_t)(n0 + lm) * DD + 8 * hf;
  const bf16_t* bv = wt + 2 * (size_t)HH * DD + (size_t)(n0 + lm) * DD + 8 * hf;

  F8 aq = {}, ak = {}, av = {};
  for (int kk = 0; kk < DD; kk += 32) {
    AF a;
    a.h[0] = *(const v8bf*)(aPtr + kk);
    a.h[1] = *(const v8bf*)(aPtr + kk + 16);
    AF fq, fk, fv;
    fq.h[0] = *(const v8bf*)(bq + kk); fq.h[1] = *(const v8bf*)(bq + kk + 16);
    fk.h[0] = *(const v8bf*)(bk + kk); fk.h[1] = *(const v8bf*)(bk + kk + 16);
    fv.h[0] = *(const v8bf*)(bv + kk); fv.h[1] = *(const v8bf*)(bv + kk + 16);
    aq.v = wmma_bf16(a.v, fq.v, aq.v);
    ak.v = wmma_bf16(a.v, fk.v, ak.v);
    av.v = wmma_bf16(a.v, fv.v, av.v);
  }

  const int bIdx = row0 / TT;
  const int t0 = row0 % TT;
#pragma unroll
  for (int r = 0; r < 8; ++r) {
    const int m = r + 8 * hf;
    const size_t tok = (size_t)(row0 + m);
    qb[tok * HH + n0 + lm] = (bf16_t)aq.f[r];
    kb[tok * HH + n0 + lm] = (bf16_t)ak.f[r];
    vt[((size_t)bIdx * HH + n0 + lm) * TT + (t0 + m)] = (bf16_t)av.f[r];
  }
}

// ---------------------------------------------------------------------------
// attn: flash attention, one wave per 16-query tile, 32-key tiles.
// With TDM: k (32x64) and v^T (64x32) tiles are DMA'd into double-buffered
// LDS by the Tensor Data Mover, overlapped with compute via TENSORcnt.
// ---------------------------------------------------------------------------
#define AW 2  // waves per block in attn

struct WaveLds {
  bf16_t k[2][32 * 64];  // [buf][key][h]   4 KB each
  bf16_t v[2][64 * 32];  // [buf][h][key]   4 KB each
  bf16_t p[16 * 32];     // P staging       1 KB
};

__global__ void __launch_bounds__(AW * 32)
attn_kernel(const bf16_t* __restrict__ qb, const bf16_t* __restrict__ kb,
            const bf16_t* __restrict__ vt, float* __restrict__ out) {
  __shared__ WaveLds smem[AW];
  const int wave = threadIdx.x >> 5;
  const int lane = threadIdx.x & 31;
  const int hf = lane >> 4;
  const int lm = lane & 15;
  const int task = blockIdx.x * AW + wave;  // 0 .. B*T/16-1
  const int b = task >> 7;                  // T/16 = 128 q-tiles per batch
  const int q0 = (task & 127) * 16;

  const bf16_t* qB = qb + (size_t)b * TT * HH;
  const bf16_t* kB = kb + (size_t)b * TT * HH;
  const bf16_t* vB = vt + (size_t)b * HH * TT;
  bf16_t* myP = smem[wave].p;

  // q A-fragments for the two H-steps; live for the whole loop.
  AF qa[2];
  {
    const bf16_t* qr = qB + (size_t)(q0 + lm) * HH + 8 * hf;
#pragma unroll
    for (int s = 0; s < 2; ++s) {
      qa[s].h[0] = *(const v8bf*)(qr + 32 * s);
      qa[s].h[1] = *(const v8bf*)(qr + 32 * s + 16);
    }
  }

  F8 o[4] = {{}, {}, {}, {}};
  float rmax[8], rsum[8];
#pragma unroll
  for (int r = 0; r < 8; ++r) { rmax[r] = -1e30f; rsum[r] = 0.0f; }

  const float scale = 0.125f;  // 64^-0.5
  const int nTiles = (q0 + 15) / 32 + 1;

#ifdef HAVE_TDM
  u32 kOff[2], vOff[2];
#pragma unroll
  for (int bfi = 0; bfi < 2; ++bfi) {
    kOff[bfi] = (u32)(size_t)(void*)&smem[wave].k[bfi][0];
    vOff[bfi] = (u32)(size_t)(void*)&smem[wave].v[bfi][0];
  }
  // prologue: DMA tile 0 into buffer 0
  tdm_load_2d(kOff[0], kB, /*td0=*/HH, /*td1=*/TT, /*tile0=*/HH, /*tile1=*/32, HH);
  tdm_load_2d(vOff[0], vB, /*td0=*/TT, /*td1=*/HH, /*tile0=*/32, /*tile1=*/HH, TT);
#endif

  for (int tile = 0; tile < nTiles; ++tile) {
    const int t0 = tile * 32;

#ifdef HAVE_TDM
    if (tile + 1 < nTiles) {
      const int tn = (tile + 1) * 32;
      const int nb = (tile + 1) & 1;
      tdm_load_2d(kOff[nb], kB + (size_t)tn * HH, HH, TT, HH, 32, HH);
      tdm_load_2d(vOff[nb], vB + tn, TT, HH, 32, HH, TT);
      __builtin_amdgcn_s_wait_tensorcnt(2);  // current tile's 2 DMAs done
    } else {
      __builtin_amdgcn_s_wait_tensorcnt(0);
    }
    asm volatile("" ::: "memory");
    const bf16_t* kTile = smem[wave].k[tile & 1];
    const bf16_t* vTile = smem[wave].v[tile & 1];
#endif

    // k^T B-fragments (K = H, N = 16 keys)
    AF kf[2][2];
#pragma unroll
    for (int s = 0; s < 2; ++s)
#pragma unroll
      for (int c = 0; c < 2; ++c) {
#ifdef HAVE_TDM
        const bf16_t* kp = kTile + (16 * s + lm) * HH + 32 * c + 8 * hf;
#else
        const bf16_t* kp = kB + (size_t)(t0 + 16 * s + lm) * HH + 32 * c + 8 * hf;
#endif
        kf[s][c].h[0] = *(const v8bf*)kp;
        kf[s][c].h[1] = *(const v8bf*)(kp + 16);
      }

    // S = q @ k^T over H=64
    F8 S[2];
#pragma unroll
    for (int s = 0; s < 2; ++s) {
      v8f z = {};
      S[s].v = wmma_bf16(qa[1].v, kf[s][1].v, wmma_bf16(qa[0].v, kf[s][0].v, z));
    }

    // scale + causal mask (row = q0 + r + 8*hf, col = t0 + 16*s + lm)
    float P[2][8];
#pragma unroll
    for (int s = 0; s < 2; ++s) {
      const int col = t0 + 16 * s + lm;
#pragma unroll
      for (int r = 0; r < 8; ++r) {
        const int rowg = q0 + r + 8 * hf;
        const float sv = S[s].f[r] * scale;
        P[s][r] = (col > rowg) ? -1e30f : sv;
      }
    }

    // online softmax: row reductions across 16 lanes of each half-wave
    float corr[8];
#pragma unroll
    for (int r = 0; r < 8; ++r) {
      float m2 = fmaxf(P[0][r], P[1][r]);
#pragma unroll
      for (int off = 1; off < 16; off <<= 1)
        m2 = fmaxf(m2, __shfl_xor(m2, off, 32));
      const float nm = fmaxf(rmax[r], m2);
      corr[r] = __expf(rmax[r] - nm);
      rmax[r] = nm;
      const float e0 = __expf(P[0][r] - nm);
      const float e1 = __expf(P[1][r] - nm);
      P[0][r] = e0; P[1][r] = e1;
      float sadd = e0 + e1;
#pragma unroll
      for (int off = 1; off < 16; off <<= 1)
        sadd += __shfl_xor(sadd, off, 32);
      rsum[r] = rsum[r] * corr[r] + sadd;
    }
#pragma unroll
    for (int nt = 0; nt < 4; ++nt)
#pragma unroll
      for (int r = 0; r < 8; ++r) o[nt].f[r] *= corr[r];

    // C layout -> A layout through wave-private LDS tile
#pragma unroll
    for (int s = 0; s < 2; ++s)
#pragma unroll
      for (int r = 0; r < 8; ++r)
        myP[(r + 8 * hf) * 32 + 16 * s + lm] = (bf16_t)P[s][r];
    asm volatile("s_wait_dscnt 0" ::: "memory");
    AF pa;
    {
      const bf16_t* pr = myP + lm * 32 + 8 * hf;
      pa.h[0] = *(const v8bf*)pr;
      pa.h[1] = *(const v8bf*)(pr + 16);
    }

    // O += P @ v
#pragma unroll
    for (int nt = 0; nt < 4; ++nt) {
      AF vf;
#ifdef HAVE_TDM
      const bf16_t* vp = vTile + (16 * nt + lm) * 32 + 8 * hf;
#else
      const bf16_t* vp = vB + (size_t)(16 * nt + lm) * TT + t0 + 8 * hf;
#endif
      vf.h[0] = *(const v8bf*)vp;
      vf.h[1] = *(const v8bf*)(vp + 16);
      o[nt].v = wmma_bf16(pa.v, vf.v, o[nt].v);
    }
  }

  // normalize and store fp32 output [B,T,H]
#pragma unroll
  for (int r = 0; r < 8; ++r) {
    const float inv = 1.0f / rsum[r];
    const int rowg = q0 + r + 8 * hf;
#pragma unroll
    for (int nt = 0; nt < 4; ++nt)
      out[((size_t)b * TT + rowg) * HH + 16 * nt + lm] = o[nt].f[r] * inv;
  }
}

// ---------------------------------------------------------------------------
extern "C" void kernel_launch(void* const* d_in, const int* in_sizes, int n_in,
                              void* d_out, int out_size, void* d_ws, size_t ws_size,
                              hipStream_t stream) {
  (void)in_sizes; (void)n_in; (void)out_size; (void)ws_size;
  const float* x  = (const float*)d_in[0];
  const float* Wq = (const float*)d_in[1];
  const float* Wk = (const float*)d_in[2];
  const float* Wv = (const float*)d_in[3];
  float* out = (float*)d_out;

  char* ws = (char*)d_ws;
  size_t off = 0;
  bf16_t* xb = (bf16_t*)(ws + off); off += (size_t)BB * TT * DD * sizeof(bf16_t); // 32 MB
  bf16_t* wt = (bf16_t*)(ws + off); off += 3ull * HH * DD * sizeof(bf16_t);       // 384 KB
  bf16_t* qb = (bf16_t*)(ws + off); off += (size_t)BB * TT * HH * sizeof(bf16_t); // 2 MB
  bf16_t* kb = (bf16_t*)(ws + off); off += (size_t)BB * TT * HH * sizeof(bf16_t); // 2 MB
  bf16_t* vt = (bf16_t*)(ws + off); off += (size_t)BB * TT * HH * sizeof(bf16_t); // 2 MB

  prep_kernel<<<2048, 256, 0, stream>>>(x, Wq, Wk, Wv, xb, wt);
  proj_kernel<<<BB * TT / 16, 128, 0, stream>>>(xb, wt, qb, kb, vt);
  attn_kernel<<<BB * TT / 16 / AW, AW * 32, 0, stream>>>(qb, kb, vt, out);
}
